// SelfAttention_77893526880376
// MI455X (gfx1250) — matrix operand order repaired and checked by reading, not compiled, not run on hardware
//
#include <hip/hip_runtime.h>
#include <hip/hip_bf16.h>

typedef __attribute__((ext_vector_type(16))) __bf16 v16bf;
typedef __attribute__((ext_vector_type(8)))  float  v8f;
typedef __attribute__((ext_vector_type(4)))  unsigned int u32x4;
typedef __attribute__((ext_vector_type(8)))  int i32x8;
typedef __attribute__((ext_vector_type(4)))  int i32x4;

union FragB {
  v16bf v;
  uint4 q[2];
};

#define LSTR 40  // LDS row stride in halfwords (32 data + 8 pad = 80B, 16B aligned)

#if __has_builtin(__builtin_amdgcn_tensor_load_to_lds)
#define HAVE_TDM 1
#else
#define HAVE_TDM 0
#endif

__device__ __forceinline__ unsigned short f2bf(float f) {
  unsigned int u = __float_as_uint(f);
  unsigned int r = u + 0x7fffu + ((u >> 16) & 1u);  // round-to-nearest-even
  return (unsigned short)(r >> 16);
}
__device__ __forceinline__ float b2f(unsigned short h) {
  return __uint_as_float(((unsigned int)h) << 16);
}
__device__ __forceinline__ v8f wmma_bf16(v16bf a, v16bf b, v8f c) {
  return __builtin_amdgcn_wmma_f32_16x16x32_bf16(false, a, false, b, (short)0, c,
                                                 false, false);
}

#if HAVE_TDM
// TDM: DMA a 2D bf16 tile [tile_rows x 32 elems] from global into LDS with a
// hardware-inserted 16B pad after every 64B row (-> LSTR=40 halfword rows).
// D# per CDNA5 ISA ch.8: group0 = {flags, lds_addr, global_addr, type=2},
// group1 = {data_size/pad, tensor dims, tile dims, strides}; groups 2/3 = 0 (2D).
__device__ __forceinline__ void tdm_load_tile_2d(const unsigned short* gptr,
                                                 unsigned int lds_off,
                                                 unsigned int stride_elems,
                                                 unsigned int rows_avail,
                                                 unsigned int tile_rows) {
  unsigned long long ga = (unsigned long long)gptr;
  u32x4 g0;
  g0[0] = 1u;                                  // count=1 (valid user descriptor)
  g0[1] = lds_off;                             // lds_addr (bytes)
  g0[2] = (unsigned int)(ga & 0xffffffffu);    // global_addr[31:0]
  g0[3] = (unsigned int)((ga >> 32) & 0x1ffffffu) | (2u << 30);  // [56:32]|type=2
  i32x8 g1;
  unsigned int td0 = stride_elems;             // tensor_dim0 (elems in a row)
  unsigned int td1 = rows_avail;               // tensor_dim1 (rows)
  g1[0] = (int)((1u << 16)      // data_size = 2 bytes
              | (1u << 20)      // pad_enable
              | (3u << 22)      // pad_interval: 16 DWORDs (64B)
              | (3u << 25));    // pad_amount:   4 DWORDs (16B)
  g1[1] = (int)((td0 & 0xffffu) << 16);
  g1[2] = (int)(((td0 >> 16) & 0xffffu) | ((td1 & 0xffffu) << 16));
  g1[3] = (int)(((td1 >> 16) & 0xffffu) | (32u << 16));  // tile_dim0 = 32 elems
  g1[4] = (int)(tile_rows & 0xffffu);                    // tile_dim1; tile_dim2=0
  g1[5] = (int)(stride_elems);                           // tensor_dim0_stride lo
  g1[6] = 0;                                             // stride hi | dim1_stride
  g1[7] = 0;
  i32x4 gz = {0, 0, 0, 0};
#if defined(__clang_major__) && __clang_major__ >= 23
  i32x8 gz8 = {0, 0, 0, 0, 0, 0, 0, 0};
  __builtin_amdgcn_tensor_load_to_lds(g0, g1, gz, gz, gz8, 0);
#else
  __builtin_amdgcn_tensor_load_to_lds(g0, g1, gz, gz, 0);
#endif
}
#endif

// ---------------------------------------------------------------- f32 -> bf16
__global__ void cvt_bf16(const float* __restrict__ src,
                         unsigned short* __restrict__ dst, int n) {
  int i = blockIdx.x * blockDim.x + threadIdx.x;
  if (i < n) dst[i] = f2bf(src[i]);
}

// ------------------------------------------------- C[M,N] = A[M,K] * B[N,K]^T
// A, B bf16 row-major over K. out_is_bf16=1: bf16 out; 0: f32 out + bias.
__global__ __launch_bounds__(256) void gemm_bf16_nt(
    const unsigned short* __restrict__ A, const unsigned short* __restrict__ B,
    void* __restrict__ Cout, const float* __restrict__ bias, int M, int N,
    int K, int out_is_bf16) {
  // double-buffered tiles: buffer layout [buf0: A,B][buf1: A,B], each 128*LSTR
  __shared__ __align__(16) unsigned short smem[4 * 128 * LSTR];

  const int tid  = threadIdx.x;
  const int lane = tid & 31;
  const int wave = tid >> 5;
  const int hf   = lane >> 4;
  const int l15  = lane & 15;
  const int wm   = (wave >> 2) * 64;
  const int wn   = (wave & 3) * 32;
  const int m0 = blockIdx.x * 128;
  const int n0 = blockIdx.y * 128;

  v8f acc[4][2];
#pragma unroll
  for (int i = 0; i < 4; ++i)
#pragma unroll
    for (int j = 0; j < 2; ++j) acc[i][j] = {};

#if HAVE_TDM
  const unsigned int lds_base = (unsigned int)(size_t)(void*)smem;
  if (wave == 0) {
    tdm_load_tile_2d(A + (size_t)m0 * K, lds_base + 0, K, M - m0, 128);
    tdm_load_tile_2d(B + (size_t)n0 * K, lds_base + 128 * LSTR * 2, K, N - n0,
                     128);
  }
#else
  const int lr = tid >> 1;
  const int lc = (tid & 1) * 16;
  const unsigned short* gA = A + (size_t)(m0 + lr) * K + lc;
  const unsigned short* gB = B + (size_t)(n0 + lr) * K + lc;
#endif

  for (int k0 = 0; k0 < K; k0 += 32) {
    const int cur = (k0 >> 5) & 1;
    unsigned short* As = smem + cur * (2 * 128 * LSTR);
    unsigned short* Bs = As + 128 * LSTR;
#if HAVE_TDM
    if (wave == 0) __builtin_amdgcn_s_wait_tensorcnt(0);
    __syncthreads();  // DMA for buf[cur] done & prev reads of buf[cur^1] done
    if (wave == 0 && k0 + 32 < K) {
      unsigned int nxt = lds_base + (unsigned)(cur ^ 1) * (2 * 128 * LSTR * 2);
      tdm_load_tile_2d(A + (size_t)m0 * K + k0 + 32, nxt, K, M - m0, 128);
      tdm_load_tile_2d(B + (size_t)n0 * K + k0 + 32, nxt + 128 * LSTR * 2, K,
                       N - n0, 128);
    }
#else
    uint4 a0 = *(const uint4*)(gA + k0);
    uint4 a1 = *(const uint4*)(gA + k0 + 8);
    uint4 b0 = *(const uint4*)(gB + k0);
    uint4 b1 = *(const uint4*)(gB + k0 + 8);
    if (k0 + 32 < K) {
      __builtin_prefetch(gA + k0 + 32, 0, 0);
      __builtin_prefetch(gB + k0 + 32, 0, 0);
    }
    __syncthreads();
    *(uint4*)(As + lr * LSTR + lc)     = a0;
    *(uint4*)(As + lr * LSTR + lc + 8) = a1;
    *(uint4*)(Bs + lr * LSTR + lc)     = b0;
    *(uint4*)(Bs + lr * LSTR + lc + 8) = b1;
    __syncthreads();
#endif

    // A frag: lane M=l15, K chunks [hf*8..+7] and [16+hf*8..+7]
    FragB af[4];
#pragma unroll
    for (int mi = 0; mi < 4; ++mi) {
      const unsigned short* rp = As + (wm + mi * 16 + l15) * LSTR;
      af[mi].q[0] = *(const uint4*)(rp + hf * 8);
      af[mi].q[1] = *(const uint4*)(rp + 16 + hf * 8);
    }
    // B frag: lane N=l15, K chunks [hf*8..+7] and [16+hf*8..+7]
    FragB bf2[2];
#pragma unroll
    for (int nj = 0; nj < 2; ++nj) {
      const unsigned short* rp = Bs + (wn + nj * 16 + l15) * LSTR;
      bf2[nj].q[0] = *(const uint4*)(rp + hf * 8);
      bf2[nj].q[1] = *(const uint4*)(rp + 16 + hf * 8);
    }
#pragma unroll
    for (int mi = 0; mi < 4; ++mi)
#pragma unroll
      for (int nj = 0; nj < 2; ++nj)
        acc[mi][nj] = wmma_bf16(af[mi].v, bf2[nj].v, acc[mi][nj]);
  }

  // C layout: lane n = l15, VGPR r -> m = r + 8*hf
#pragma unroll
  for (int mi = 0; mi < 4; ++mi)
#pragma unroll
    for (int nj = 0; nj < 2; ++nj)
#pragma unroll
      for (int r = 0; r < 8; ++r) {
        int gm = m0 + wm + mi * 16 + r + 8 * hf;
        int gn = n0 + wn + nj * 16 + l15;
        float v = acc[mi][nj][r];
        if (out_is_bf16)
          ((unsigned short*)Cout)[(size_t)gm * N + gn] = f2bf(v);
        else
          ((float*)Cout)[(size_t)gm * N + gn] = v + bias[gn];
      }
}

// ---------------- per-head LayerNorm on K,Q (+ E^-0.25 scale) and V transpose
// inputs  [B,T,H*64] bf16 ; Kn/Qn -> [B,H,T,64] bf16 ; Vt -> [B,H,64,T] bf16
__global__ __launch_bounds__(256) void ln_heads(
    const unsigned short* __restrict__ Kp, const unsigned short* __restrict__ Qp,
    const unsigned short* __restrict__ Vp, const float* __restrict__ kg,
    const float* __restrict__ kbi, const float* __restrict__ qg,
    const float* __restrict__ qbi, unsigned short* __restrict__ Kn,
    unsigned short* __restrict__ Qn, unsigned short* __restrict__ Vt, int Bn,
    int Tn, int Hn) {
  const int lane = threadIdx.x & 31;
  int row = blockIdx.x * 8 + (threadIdx.x >> 5);  // one wave per (b,t,h)
  if (row >= Bn * Tn * Hn) return;
  int h = row % Hn;
  int t = (row / Hn) % Tn;
  int b = row / (Hn * Tn);
  const float scale = 0.17677669529663687f;  // 1024^(-1/4)

  size_t src = ((size_t)(b * Tn + t)) * (Hn * 64) + h * 64;
  size_t dst = (((size_t)(b * Hn + h)) * Tn + t) * 64;

#pragma unroll
  for (int which = 0; which < 2; ++which) {
    const unsigned short* in = which ? Qp : Kp;
    const float* g = which ? qg : kg;
    const float* bb = which ? qbi : kbi;
    unsigned short* out = which ? Qn : Kn;
    float x0 = b2f(in[src + lane]);
    float x1 = b2f(in[src + 32 + lane]);
    float s = x0 + x1;
#pragma unroll
    for (int m = 1; m < 32; m <<= 1) s += __shfl_xor(s, m, 32);
    float mean = s * (1.0f / 64.0f);
    float d0 = x0 - mean, d1 = x1 - mean;
    float vs = d0 * d0 + d1 * d1;
#pragma unroll
    for (int m = 1; m < 32; m <<= 1) vs += __shfl_xor(vs, m, 32);
    float rinv = rsqrtf(vs * (1.0f / 64.0f) + 1e-5f);
    out[dst + lane]      = f2bf((d0 * rinv * g[lane] + bb[lane]) * scale);
    out[dst + 32 + lane] = f2bf((d1 * rinv * g[lane + 32] + bb[lane + 32]) * scale);
  }
  size_t vdst = (((size_t)(b * Hn + h)) * 64 + lane) * Tn + t;
  Vt[vdst]                   = Vp[src + lane];
  Vt[vdst + 32 * (size_t)Tn] = Vp[src + 32 + lane];
}

// ----------------------------------------- flash attention, 16 q-rows / wave
__global__ __launch_bounds__(256) void attn_fwd(
    const unsigned short* __restrict__ Qn, const unsigned short* __restrict__ Kn,
    const unsigned short* __restrict__ Vt, unsigned short* __restrict__ Ob,
    int Hn, int Tn) {
  __shared__ __align__(16) unsigned short Pt[8][16 * LSTR];
  const int lane = threadIdx.x & 31;
  const int wave = threadIdx.x >> 5;
  const int hf   = lane >> 4;
  const int l15  = lane & 15;
  const int bh = blockIdx.x;
  const int b = bh / Hn;
  const int h = bh % Hn;
  const int q0 = blockIdx.y * 128 + wave * 16;

  const unsigned short* Qh = Qn + (size_t)bh * Tn * 64;
  const unsigned short* Kh = Kn + (size_t)bh * Tn * 64;
  const unsigned short* Vh = Vt + (size_t)bh * 64 * Tn;

  FragB qa[2];  // A fragments for d=[0,32) and [32,64)
#pragma unroll
  for (int dh = 0; dh < 2; ++dh) {
    const unsigned short* rp = Qh + (size_t)(q0 + l15) * 64 + dh * 32;
    qa[dh].q[0] = *(const uint4*)(rp + hf * 8);
    qa[dh].q[1] = *(const uint4*)(rp + 16 + hf * 8);
  }

  v8f o[4];
#pragma unroll
  for (int i = 0; i < 4; ++i) o[i] = {};
  float mrow[8], lrow[8];
#pragma unroll
  for (int r = 0; r < 8; ++r) { mrow[r] = -3.0e38f; lrow[r] = 0.0f; }

  unsigned short* myP = &Pt[wave][0];

  for (int jb = 0; jb < q0 + 16; jb += 32) {  // causal: only blocks with keys<=q
    v8f s[2];
#pragma unroll
    for (int x = 0; x < 2; ++x) {
      s[x] = {};
#pragma unroll
      for (int dh = 0; dh < 2; ++dh) {
        FragB kf;  // B frag: lane n = key l15, d chunks [hf*8..+7], [16+hf*8..+7]
        const unsigned short* rp =
            Kh + (size_t)(jb + x * 16 + l15) * 64 + dh * 32;
        kf.q[0] = *(const uint4*)(rp + hf * 8);
        kf.q[1] = *(const uint4*)(rp + 16 + hf * 8);
        s[x] = wmma_bf16(qa[dh].v, kf.v, s[x]);
      }
    }
    // online softmax over 32 keys; row m = r + 8*hf, col = key = l15 (+16)
#pragma unroll
    for (int r = 0; r < 8; ++r) {
      const int qpos = q0 + r + 8 * hf;
      float s0 = (jb + l15 <= qpos) ? s[0][r] : -3.0e38f;
      float s1 = (jb + 16 + l15 <= qpos) ? s[1][r] : -3.0e38f;
      float mx = fmaxf(s0, s1);
#pragma unroll
      for (int m = 1; m < 16; m <<= 1) mx = fmaxf(mx, __shfl_xor(mx, m, 32));
      float mn = fmaxf(mrow[r], mx);
      float corr = __expf(mrow[r] - mn);
      float p0 = __expf(s0 - mn);
      float p1 = __expf(s1 - mn);
      float ps = p0 + p1;
#pragma unroll
      for (int m = 1; m < 16; m <<= 1) ps += __shfl_xor(ps, m, 32);
      lrow[r] = lrow[r] * corr + ps;
      mrow[r] = mn;
#pragma unroll
      for (int d = 0; d < 4; ++d) o[d][r] *= corr;
      myP[(r + 8 * hf) * LSTR + l15]      = f2bf(p0);
      myP[(r + 8 * hf) * LSTR + 16 + l15] = f2bf(p1);
    }
    // wave-local LDS bounce: C-layout -> A-fragment layout (DS ops are
    // in-order per wave; drain DS counter, no barrier needed)
    asm volatile("s_wait_dscnt 0" ::: "memory");
    FragB pf;
    {
      const unsigned short* rp = myP + l15 * LSTR;
      pf.q[0] = *(const uint4*)(rp + hf * 8);
      pf.q[1] = *(const uint4*)(rp + 16 + hf * 8);
    }
#pragma unroll
    for (int d = 0; d < 4; ++d) {
      FragB vf;  // B frag from V^T: lane n = d-col, keys [hf*8..+7], [16+hf*8..+7]
      const unsigned short* rp =
          Vh + (size_t)(d * 16 + l15) * Tn + jb;
      vf.q[0] = *(const uint4*)(rp + hf * 8);
      vf.q[1] = *(const uint4*)(rp + 16 + hf * 8);
      o[d] = wmma_bf16(pf.v, vf.v, o[d]);
    }
  }

#pragma unroll
  for (int r = 0; r < 8; ++r) {
    float inv = 1.0f / lrow[r];
#pragma unroll
    for (int d = 0; d < 4; ++d) {
      size_t idx = ((size_t)(b * Tn + q0 + r + 8 * hf)) * (Hn * 64) + h * 64 +
                   d * 16 + l15;
      Ob[idx] = f2bf(o[d][r] * inv);
    }
  }
}

// ---------------------------------------------------------------------------
extern "C" void kernel_launch(void* const* d_in, const int* in_sizes, int n_in,
                              void* d_out, int out_size, void* d_ws,
                              size_t ws_size, hipStream_t stream) {
  (void)n_in; (void)out_size; (void)ws_size;
  const int E = 1024, H = 16, T = 2048;
  const float* x    = (const float*)d_in[0];
  const float* Wk   = (const float*)d_in[1];
  const float* Wq   = (const float*)d_in[2];
  const float* Wv   = (const float*)d_in[3];
  const float* Wo   = (const float*)d_in[4];
  const float* bo   = (const float*)d_in[5];
  const float* klng = (const float*)d_in[6];
  const float* klnb = (const float*)d_in[7];
  const float* qlng = (const float*)d_in[8];
  const float* qlnb = (const float*)d_in[9];

  const int M = in_sizes[0] / E;  // B*T = 8192
  const int Bn = M / T;           // 4

  char* ws = (char*)d_ws;
  size_t off = 0;
  auto alloc = [&](size_t bytes) {
    char* p = ws + off;
    off += (bytes + 255) & ~(size_t)255;
    return p;
  };
  unsigned short* xb   = (unsigned short*)alloc((size_t)M * E * 2);
  unsigned short* wkb  = (unsigned short*)alloc((size_t)E * E * 2);
  unsigned short* wqb  = (unsigned short*)alloc((size_t)E * E * 2);
  unsigned short* wvb  = (unsigned short*)alloc((size_t)E * E * 2);
  unsigned short* wob  = (unsigned short*)alloc((size_t)E * E * 2);
  unsigned short* Kp   = (unsigned short*)alloc((size_t)M * E * 2);
  unsigned short* Qp   = (unsigned short*)alloc((size_t)M * E * 2);
  unsigned short* Vp   = (unsigned short*)alloc((size_t)M * E * 2);
  unsigned short* Kn   = (unsigned short*)alloc((size_t)M * E * 2);
  unsigned short* Qn   = (unsigned short*)alloc((size_t)M * E * 2);
  unsigned short* Vt   = (unsigned short*)alloc((size_t)M * E * 2);
  unsigned short* attb = (unsigned short*)alloc((size_t)M * E * 2);

  int nx = M * E;
  cvt_bf16<<<(nx + 255) / 256, 256, 0, stream>>>(x, xb, nx);
  int nw = E * E;
  cvt_bf16<<<(nw + 255) / 256, 256, 0, stream>>>(Wk, wkb, nw);
  cvt_bf16<<<(nw + 255) / 256, 256, 0, stream>>>(Wq, wqb, nw);
  cvt_bf16<<<(nw + 255) / 256, 256, 0, stream>>>(Wv, wvb, nw);
  cvt_bf16<<<(nw + 255) / 256, 256, 0, stream>>>(Wo, wob, nw);

  dim3 g(M / 128, E / 128);
  gemm_bf16_nt<<<g, 256, 0, stream>>>(xb, wkb, Kp, nullptr, M, E, E, 1);
  gemm_bf16_nt<<<g, 256, 0, stream>>>(xb, wqb, Qp, nullptr, M, E, E, 1);
  gemm_bf16_nt<<<g, 256, 0, stream>>>(xb, wvb, Vp, nullptr, M, E, E, 1);

  int rows = Bn * T * H;
  ln_heads<<<(rows + 7) / 8, 256, 0, stream>>>(Kp, Qp, Vp, klng, klnb, qlng,
                                               qlnb, Kn, Qn, Vt, Bn, T, H);

  dim3 ga(Bn * H, T / 128);
  attn_fwd<<<ga, 256, 0, stream>>>(Qn, Kn, Vt, attb, H, T);

  gemm_bf16_nt<<<g, 256, 0, stream>>>(attb, wob, d_out, bo, M, E, E, 0);
}
